// GraphAttentionLayer_77549929497111
// MI455X (gfx1250) — matrix operand order repaired and checked by reading, not compiled, not run on hardware
//
#include <hip/hip_runtime.h>
#include <hip/hip_bf16.h>

typedef _Float16 v16h __attribute__((ext_vector_type(16)));
typedef float    v8f  __attribute__((ext_vector_type(8)));

#define NROWS   8192
#define INDIM   256
#define OUTDIM  64
#define NHEADS  8
#define NWAVES  8   // waves per workgroup in the fused kernel

// workspace layout (bytes)
#define WS_TMAX 0
#define WS_S    1024
#define WS_T    (WS_S + NROWS * 4)
#define WS_WH   (WS_T + NROWS * 4)                 // fp32 Wh (8192x64)
#define WS_WHT  (WS_WH + NROWS * OUTDIM * 4)       // f16  Wh^T (64x8192)

// monotonic float->uint key for deterministic atomic max
__device__ __forceinline__ unsigned fkey(float f) {
  unsigned b = __float_as_uint(f);
  return (b & 0x80000000u) ? ~b : (b | 0x80000000u);
}
__device__ __forceinline__ float fdecode(unsigned k) {
  return __uint_as_float((k & 0x80000000u) ? (k ^ 0x80000000u) : ~k);
}

__global__ void k_init(unsigned* __restrict__ tmax_key) { *tmax_key = 0u; }

// ---------------------------------------------------------------------------
// Kernel 1: Wh = X(8192x256) @ W(256x64) via v_wmma_f32_16x16x32_f16.
// One wave computes a 16x64 output strip (4 C tiles). Also emits WhT (f16).
// ---------------------------------------------------------------------------
__global__ __launch_bounds__(256) void k_gemm1(const float* __restrict__ X,
                                               const float* __restrict__ W,
                                               float* __restrict__ Wh,
                                               _Float16* __restrict__ WhT) {
  const int lane = threadIdx.x & 31;
  const int wave = threadIdx.x >> 5;
  const int l15  = lane & 15;
  const int h    = lane >> 4;            // which half of the wave
  const int rb   = blockIdx.x * 8 + wave;
  const int i0   = rb * 16;
  const int i    = i0 + l15;             // A-matrix row held by this lane

  const float* xrow = X + (size_t)i * INDIM;
  v8f c[4] = {};

  for (int kb = 0; kb < INDIM / 32; ++kb) {
    const int k0 = kb * 32;
    // ---- A operand: 16x32 f16 tile. Lane layout per ISA 7.12.2:
    // elems 0..7 -> K = h*8 + kk ; elems 8..15 -> K = 16 + h*8 + (kk-8)
    float4 x0 = *(const float4*)(xrow + k0 + h * 8);
    float4 x1 = *(const float4*)(xrow + k0 + h * 8 + 4);
    float4 x2 = *(const float4*)(xrow + k0 + 16 + h * 8);
    float4 x3 = *(const float4*)(xrow + k0 + 16 + h * 8 + 4);
    float xa[16];
    *(float4*)(xa + 0)  = x0; *(float4*)(xa + 4)  = x1;
    *(float4*)(xa + 8)  = x2; *(float4*)(xa + 12) = x3;
    v16h a;
#pragma unroll
    for (int kk = 0; kk < 16; ++kk) a[kk] = (_Float16)xa[kk];

    // ---- B operands: 32x16 f16 tiles. elems 0..7 -> K = k0+h*8+kk ; elems 8..15 -> K = k0+16+h*8+(kk-8)
    v16h b0, b1, b2, b3;
#pragma unroll
    for (int kk = 0; kk < 16; ++kk) {
      const float* wr = W + (size_t)(k0 + h * 8 + kk + (kk >= 8 ? 8 : 0)) * OUTDIM + l15;
      b0[kk] = (_Float16)wr[0];
      b1[kk] = (_Float16)wr[16];
      b2[kk] = (_Float16)wr[32];
      b3[kk] = (_Float16)wr[48];
    }
    c[0] = __builtin_amdgcn_wmma_f32_16x16x32_f16(false, a, false, b0, (short)0, c[0], false, false);
    c[1] = __builtin_amdgcn_wmma_f32_16x16x32_f16(false, a, false, b1, (short)0, c[1], false, false);
    c[2] = __builtin_amdgcn_wmma_f32_16x16x32_f16(false, a, false, b2, (short)0, c[2], false, false);
    c[3] = __builtin_amdgcn_wmma_f32_16x16x32_f16(false, a, false, b3, (short)0, c[3], false, false);
  }

  // C layout: VGPR r -> M = r + 8*h, N = nt*16 + l15
#pragma unroll
  for (int nt = 0; nt < 4; ++nt) {
#pragma unroll
    for (int r = 0; r < 8; ++r) {
      const int M = r + h * 8;
      const int n = nt * 16 + l15;
      const float v = c[nt][r];
      Wh[(size_t)(i0 + M) * OUTDIM + n] = v;
      WhT[(size_t)n * NROWS + i0 + M] = (_Float16)v;  // f16 transposed copy
    }
  }
}

// ---------------------------------------------------------------------------
// Kernel 2: s_i = Wh[i,:]@a[:64], t_i = Wh[i,:]@a[64:], tmax = max_i t_i
// ---------------------------------------------------------------------------
__global__ __launch_bounds__(256) void k_st(const float* __restrict__ Wh,
                                            const float* __restrict__ a,
                                            float* __restrict__ s_g,
                                            float* __restrict__ t_g,
                                            unsigned* __restrict__ tmax_key) {
  const int i = blockIdx.x * 256 + threadIdx.x;
  const float* row = Wh + (size_t)i * OUTDIM;
  float s = 0.f, t = 0.f;
#pragma unroll
  for (int n = 0; n < OUTDIM; n += 4) {
    float4 w  = *(const float4*)(row + n);
    float4 a1 = *(const float4*)(a + n);
    float4 a2 = *(const float4*)(a + OUTDIM + n);
    s += w.x * a1.x + w.y * a1.y + w.z * a1.z + w.w * a1.w;
    t += w.x * a2.x + w.y * a2.y + w.z * a2.z + w.w * a2.w;
  }
  s_g[i] = s;
  t_g[i] = t;
  atomicMax(tmax_key, fkey(t));  // commutative & idempotent -> deterministic
}

// ---------------------------------------------------------------------------
// Kernel 3 (fused GAT): per 16-row block, single pass over adj:
//   p_ij = adj * exp(leaky(s_i + t_j) - m_i),  m_i = leaky(s_i + tmax) >= max_j
//   out[i,:] = (sum_j p_ij * Wh[j,:]) / (sum_j p_ij), tiled x8 heads.
// WMMA f16 A = probabilities, B = WhT (contiguous 32B per-lane loads).
// ---------------------------------------------------------------------------
__global__ __launch_bounds__(256) void k_gat(const int* __restrict__ adj,
                                             const float* __restrict__ s_g,
                                             const float* __restrict__ t_g,
                                             const _Float16* __restrict__ WhT,
                                             const unsigned* __restrict__ tmax_key,
                                             float* __restrict__ out) {
  __shared__ float c_red[NWAVES * 16 * OUTDIM];  // 32 KB
  __shared__ float z_red[NWAVES * 32];           // 1 KB

  const int lane = threadIdx.x & 31;
  const int wave = threadIdx.x >> 5;
  const int l15  = lane & 15;
  const int h    = lane >> 4;
  const int i0   = blockIdx.x * 16;
  const int i    = i0 + l15;

  const float tmax = fdecode(*tmax_key);
  const float si   = s_g[i];
  const float mx   = si + tmax;
  const float m_i  = fmaxf(mx, 0.01f * mx);   // leaky(s_i + tmax), global row bound

  const int* adjRow = adj + (size_t)i * NROWS;
  v8f c[4] = {};
  float zpart = 0.f;

  for (int ch = wave; ch < NROWS / 32; ch += NWAVES) {
    const int j0 = ch * 32;
    __builtin_prefetch(adjRow + j0 + 32 * NWAVES, 0, 0);  // next chunk for this wave

    // adj + t for this lane's 16 (row,col) slots, matching f16 A VGPR layout
    int4   A0 = *(const int4*)(adjRow + j0 + h * 16);
    int4   A1 = *(const int4*)(adjRow + j0 + h * 16 + 4);
    int4   A2 = *(const int4*)(adjRow + j0 + h * 16 + 8);
    int4   A3 = *(const int4*)(adjRow + j0 + h * 16 + 12);
    float4 T0 = *(const float4*)(t_g + j0 + h * 16);
    float4 T1 = *(const float4*)(t_g + j0 + h * 16 + 4);
    float4 T2 = *(const float4*)(t_g + j0 + h * 16 + 8);
    float4 T3 = *(const float4*)(t_g + j0 + h * 16 + 12);
    int av[16]; float tv[16];
    *(int4*)(av + 0) = A0;  *(int4*)(av + 4) = A1;
    *(int4*)(av + 8) = A2;  *(int4*)(av + 12) = A3;
    *(float4*)(tv + 0) = T0; *(float4*)(tv + 4) = T1;
    *(float4*)(tv + 8) = T2; *(float4*)(tv + 12) = T3;

    v16h pa;
    float z = zpart;
#pragma unroll
    for (int kk = 0; kk < 16; ++kk) {
      float x = si + tv[kk];
      float l = fmaxf(x, 0.01f * x);        // leaky_relu
      float p = __expf(l - m_i);            // <= 1, no overflow
      p = (av[kk] > 0) ? p : 0.0f;          // adjacency mask
      z += p;
      pa[kk] = (_Float16)p;
    }
    zpart = z;

    // B tiles from WhT: lane holds column n = nt*16+l15, K run j0+h*16..+15 (32B)
    const _Float16* bp = WhT + (size_t)l15 * NROWS + j0 + h * 16;
    v16h b0 = *(const v16h*)(bp);
    v16h b1 = *(const v16h*)(bp + (size_t)16 * NROWS);
    v16h b2 = *(const v16h*)(bp + (size_t)32 * NROWS);
    v16h b3 = *(const v16h*)(bp + (size_t)48 * NROWS);

    c[0] = __builtin_amdgcn_wmma_f32_16x16x32_f16(false, pa, false, b0, (short)0, c[0], false, false);
    c[1] = __builtin_amdgcn_wmma_f32_16x16x32_f16(false, pa, false, b1, (short)0, c[1], false, false);
    c[2] = __builtin_amdgcn_wmma_f32_16x16x32_f16(false, pa, false, b2, (short)0, c[2], false, false);
    c[3] = __builtin_amdgcn_wmma_f32_16x16x32_f16(false, pa, false, b3, (short)0, c[3], false, false);
  }

  // dump per-wave partial C tiles + Z into LDS
#pragma unroll
  for (int nt = 0; nt < 4; ++nt) {
#pragma unroll
    for (int r = 0; r < 8; ++r) {
      const int M = r + h * 8;
      c_red[(wave * 16 + M) * OUTDIM + nt * 16 + l15] = c[nt][r];
    }
  }
  z_red[wave * 32 + lane] = zpart;
  __syncthreads();

  // cross-wave reduce, normalize, tile across 8 heads
  for (int e = threadIdx.x; e < 16 * OUTDIM; e += 256) {
    const int ii = e >> 6;
    const int n  = e & 63;
    float sum = 0.f, z = 0.f;
#pragma unroll
    for (int w = 0; w < NWAVES; ++w) {
      sum += c_red[(w * 16 + ii) * OUTDIM + n];
      z   += z_red[w * 32 + ii] + z_red[w * 32 + 16 + ii];
    }
    const float val = sum / z;
    const size_t ro = (size_t)(i0 + ii) * (OUTDIM * NHEADS) + n;
#pragma unroll
    for (int hd = 0; hd < NHEADS; ++hd) out[ro + (size_t)hd * OUTDIM] = val;
  }
}

extern "C" void kernel_launch(void* const* d_in, const int* in_sizes, int n_in,
                              void* d_out, int out_size, void* d_ws, size_t ws_size,
                              hipStream_t stream) {
  const float* X   = (const float*)d_in[0];  // (8192, 256) f32
  const int*   adj = (const int*)d_in[1];    // (8192, 8192) i32
  const float* W   = (const float*)d_in[2];  // (256, 64) f32
  const float* a   = (const float*)d_in[3];  // (128, 1) f32
  float* out = (float*)d_out;                // (8192, 512) f32

  char* ws = (char*)d_ws;
  unsigned* tmax_key = (unsigned*)(ws + WS_TMAX);
  float*    s_g      = (float*)(ws + WS_S);
  float*    t_g      = (float*)(ws + WS_T);
  float*    Wh       = (float*)(ws + WS_WH);
  _Float16* WhT      = (_Float16*)(ws + WS_WHT);

  hipLaunchKernelGGL(k_init, dim3(1), dim3(1), 0, stream, tmax_key);
  hipLaunchKernelGGL(k_gemm1, dim3(NROWS / 16 / 8), dim3(256), 0, stream, X, W, Wh, WhT);
  hipLaunchKernelGGL(k_st, dim3(NROWS / 256), dim3(256), 0, stream, Wh, a, s_g, t_g, tmax_key);
  hipLaunchKernelGGL(k_gat, dim3(NROWS / 16), dim3(256), 0, stream, adj, s_g, t_g, WhT, tmax_key, out);
}
